// BeamSearch_18889266167953
// MI455X (gfx1250) — matrix operand order repaired and checked by reading, not compiled, not run on hardware
//
#include <hip/hip_runtime.h>

// ---------------------------------------------------------------------------
// Seq2seq bi-LSTM encoder + LSTM-decoder beam search for MI455X (gfx1250).
// Matrix work runs on v_wmma_f32_16x16x32_bf16; recurrent scans are VALU+LDS.
// out_W is converted once per launch to bf16 (103 MB -> L2-resident across the
// 20 decoder steps; 23.3 TB/s HBM => ~90us for the dominant logits GEMMs).
// ---------------------------------------------------------------------------

#define VOCAB   50257
#define VPAD    50272      // 3142 tiles of 16
#define NTILV   3142
#define EDIM    1024
#define HDIM    512
#define DDIM    1024
#define SLEN    64
#define KBEAM   4
#define MAXLEN  20
#define BOS_T   1
#define EOS_T   2

typedef __attribute__((ext_vector_type(16))) __bf16 v16bf;
typedef __attribute__((ext_vector_type(8)))  __bf16 v8bf;
typedef __attribute__((ext_vector_type(8)))  float  v8f;

__device__ __forceinline__ unsigned short f2bf(float f) {
  unsigned u = __float_as_uint(f);
  u += 0x7fffu + ((u >> 16) & 1u);          // round-to-nearest-even
  return (unsigned short)(u >> 16);
}
__device__ __forceinline__ float sigm(float x) { return 1.f / (1.f + __expf(-x)); }

// ---------------------------------------------------------------------------
// One 16x16 f32 tile of C = A * W^T using V_WMMA_F32_16X16X32_BF16.
// A: bf16 row-major (lda elems/row), tile row m sources A row aRow0 + aStep*m.
// W: bf16 row-major (ldw elems/row); tile col n sources W row wRow0 + n.
// ---------------------------------------------------------------------------
__device__ v8f wmma_tile(const unsigned short* A, int lda, int aRow0, int aStep,
                         const unsigned short* W, int ldw, int wRow0, int K) {
  const int lane = threadIdx.x & 31;
  const int hi   = lane >> 4;     // K-half select
  const int l16  = lane & 15;
  v8f acc = {};
  const unsigned short* aP = A + (size_t)(aRow0 + aStep * l16) * lda + hi * 8;
  const unsigned short* wP = W + (size_t)(wRow0 + l16) * ldw + hi * 8;
  for (int k0 = 0; k0 < K; k0 += 32) {
    v8bf a0 = *(const v8bf*)(aP + k0);
    v8bf a1 = *(const v8bf*)(aP + k0 + 16);
    v16bf av;
#pragma unroll
    for (int i = 0; i < 8; ++i) { av[i] = a0[i]; av[i + 8] = a1[i]; }
    v8bf b0 = *(const v8bf*)(wP + k0);
    v8bf b1 = *(const v8bf*)(wP + k0 + 16);
    v16bf bv;
#pragma unroll
    for (int i = 0; i < 8; ++i) { bv[i] = b0[i]; bv[i + 8] = b1[i]; }
    acc = __builtin_amdgcn_wmma_f32_16x16x32_bf16(false, av, false, bv,
                                                  (short)0, acc, false, false);
  }
  return acc;
}

// ------------------------------- prep kernels ------------------------------
__global__ void k_conv(const float* s, unsigned short* d, size_t n) {
  for (size_t i = (size_t)blockIdx.x * blockDim.x + threadIdx.x; i < n;
       i += (size_t)gridDim.x * blockDim.x)
    d[i] = f2bf(s[i]);
}
__global__ void k_build_decW(const float* Wih, const float* Whh, unsigned short* d) {
  size_t n = (size_t)4096 * 2048;
  for (size_t i = (size_t)blockIdx.x * blockDim.x + threadIdx.x; i < n;
       i += (size_t)gridDim.x * blockDim.x) {
    size_t j = i >> 11, e = i & 2047;
    float v = (e < 1024) ? Wih[j * 1024 + e] : Whh[j * 1024 + (e - 1024)];
    d[i] = f2bf(v);
  }
}
__global__ void k_build_outW(const float* W, unsigned short* d) {
  size_t n = (size_t)VPAD * 1024;
  for (size_t i = (size_t)blockIdx.x * blockDim.x + threadIdx.x; i < n;
       i += (size_t)gridDim.x * blockDim.x) {
    size_t r = i >> 10, c = i & 1023;
    d[i] = (r < VOCAB) ? f2bf(W[r * 1024 + c]) : (unsigned short)0;
  }
}
__global__ void k_gather_x(const int* sentence, const float* emb, unsigned short* xw) {
  size_t n = (size_t)SLEN * EDIM;
  for (size_t i = (size_t)blockIdx.x * blockDim.x + threadIdx.x; i < n;
       i += (size_t)gridDim.x * blockDim.x) {
    size_t s = i >> 10, e = i & 1023;
    xw[i] = f2bf(emb[(size_t)sentence[s] * EDIM + e]);
  }
}

// --------------------------- encoder: Z = x @ Wih^T + b --------------------
__global__ void k_enc_gemm(const unsigned short* xw,
                           const unsigned short* WihF, const unsigned short* WihB,
                           const float* bF, const float* bB, float* Zf, float* Zb) {
  int tile = blockIdx.x * 8 + (threadIdx.x >> 5);   // 512 tiles: 4 M x 128 N
  int dir  = blockIdx.y;
  int tM = tile >> 7, tN = tile & 127;
  const unsigned short* W = dir ? WihB : WihF;
  const float* b = dir ? bB : bF;
  float* Z = dir ? Zb : Zf;
  int aRow0 = dir ? (SLEN - 1 - tM * 16) : tM * 16;  // backward scan reads x reversed
  int aStep = dir ? -1 : 1;
  v8f acc = wmma_tile(xw, EDIM, aRow0, aStep, W, EDIM, tN * 16, EDIM);
  int lane = threadIdx.x & 31, hi = lane >> 4, n = (lane & 15) + tN * 16;
  float bias = b[n];
#pragma unroll
  for (int v = 0; v < 8; ++v) {
    int m = v + 8 * hi + tM * 16;
    Z[(size_t)m * 2048 + n] = acc[v] + bias;
  }
}

// --------------- encoder recurrent scan (h in LDS, Whh from L2) ------------
__global__ void k_enc_scan(const float* Zf, const float* Zb,
                           const float* WhhF, const float* WhhB,
                           float* houtF, float* houtB) {
  __shared__ float hs[HDIM], cs[HDIM];
  int u = threadIdx.x;
  int dir = blockIdx.x;
  const float* Z   = dir ? Zb : Zf;
  const float* Whh = dir ? WhhB : WhhF;
  int steps = dir ? 1 : SLEN;     // only hb[0] (first backward step) is consumed
  hs[u] = 0.f; cs[u] = 0.f;
  __syncthreads();
  for (int t = 0; t < steps; ++t) {
    float zg4[4];
#pragma unroll
    for (int g = 0; g < 4; ++g) {
      int row = g * HDIM + u;
      const float* wr = Whh + (size_t)row * HDIM;
      float acc = Z[(size_t)t * 2048 + row];
      for (int e = 0; e < HDIM; ++e) acc += wr[e] * hs[e];
      zg4[g] = acc;
    }
    __syncthreads();
    float c = sigm(zg4[1]) * cs[u] + sigm(zg4[0]) * tanhf(zg4[2]);
    cs[u] = c;
    hs[u] = sigm(zg4[3]) * tanhf(c);
    __syncthreads();
  }
  (dir ? houtB : houtF)[u] = hs[u];
}

// ------------------------------ decoder init -------------------------------
__global__ void k_init_dec(const float* hF, const float* hB, float* h0, float* c0,
                           int* tokens, int* last, int* fin, float* probs) {
  int i = blockIdx.x * blockDim.x + threadIdx.x;
  if (i < KBEAM * DDIM) {
    int u = i & 1023;
    h0[i] = (u < HDIM) ? hF[u] : hB[u - HDIM];
    c0[i] = 0.f;
  }
  if (i < KBEAM * (MAXLEN + 1)) tokens[i] = BOS_T;
  if (i < KBEAM) { last[i] = BOS_T; fin[i] = 0; probs[i] = (i == 0) ? 1.f : 0.f; }
}

// --------------------------- decoder per-step kernels ----------------------
__global__ void k_stage_gates(const int* last, const float* dec_emb,
                              const float* h, unsigned short* Ag, float* embf) {
  int i = blockIdx.x * blockDim.x + threadIdx.x;     // 4 x 2048
  if (i >= KBEAM * 2048) return;
  int k = i >> 11, e = i & 2047;
  float v;
  if (e < 1024) { v = dec_emb[(size_t)last[k] * EDIM + e]; embf[k * 1024 + e] = v; }
  else v = h[k * 1024 + (e - 1024)];
  Ag[(size_t)k * 2048 + e] = f2bf(v);
}

__global__ void k_gemm_gates(const unsigned short* Ag, const unsigned short* decW,
                             const float* bias, float* z) {
  int tile = blockIdx.x * 8 + (threadIdx.x >> 5);    // 256 N tiles
  v8f acc = wmma_tile(Ag, 2048, 0, 1, decW, 2048, tile * 16, 2048);
  int lane = threadIdx.x & 31, hi = lane >> 4, n = (lane & 15) + tile * 16;
  if (hi == 0) {
    float b = bias[n];
#pragma unroll
    for (int v = 0; v < KBEAM; ++v) z[(size_t)v * 4096 + n] = acc[v] + b;
  }
}

__global__ void k_dec_cell(const float* z, const float* c_in, float* h_cell, float* c_cell) {
  int i = blockIdx.x * blockDim.x + threadIdx.x;     // 4 x 1024
  if (i >= KBEAM * DDIM) return;
  int k = i >> 10, u = i & 1023;
  const float* zr = z + (size_t)k * 4096;
  float c = sigm(zr[1024 + u]) * c_in[i] + sigm(zr[u]) * tanhf(zr[2048 + u]);
  c_cell[i] = c;
  h_cell[i] = sigm(zr[3072 + u]) * tanhf(c);
}

__global__ void k_stage_feat(const float* h_cell, const float* embf, unsigned short* Af) {
  int i = blockIdx.x * blockDim.x + threadIdx.x;
  if (i >= KBEAM * 2048) return;
  int k = i >> 11, e = i & 2047;
  float v = (e < 1024) ? h_cell[k * 1024 + e] : embf[k * 1024 + (e - 1024)];
  Af[(size_t)k * 2048 + e] = f2bf(v);
}

__global__ void k_gemm_cat(const unsigned short* Af, const unsigned short* catW,
                           const float* bias, unsigned short* Aout) {
  int tile = blockIdx.x * 8 + (threadIdx.x >> 5);    // 64 N tiles
  v8f acc = wmma_tile(Af, 2048, 0, 1, catW, 2048, tile * 16, 2048);
  int lane = threadIdx.x & 31, hi = lane >> 4, n = (lane & 15) + tile * 16;
  if (hi == 0) {
    float b = bias[n];
#pragma unroll
    for (int v = 0; v < KBEAM; ++v)
      Aout[(size_t)v * 1024 + n] = f2bf(tanhf(acc[v] + b));
  }
}

__global__ void k_gemm_logits(const unsigned short* Aout, const unsigned short* outW,
                              const float* out_b, float* logits) {
  int tile = blockIdx.x * 8 + (threadIdx.x >> 5);
  if (tile >= NTILV) return;                          // wave-uniform exit
  v8f acc = wmma_tile(Aout, 1024, 0, 1, outW, 1024, tile * 16, 1024);
  int lane = threadIdx.x & 31, hi = lane >> 4, n = (lane & 15) + tile * 16;
  if (hi == 0) {
    bool ok = (n < VOCAB);
    float b = ok ? out_b[n] : 0.f;
#pragma unroll
    for (int v = 0; v < KBEAM; ++v)
      logits[(size_t)v * VPAD + n] = ok ? (acc[v] + b) : -1e30f;
  }
}

__global__ void k_softmax_red(const float* logits, float* rowmax, float* rowsum) {
  __shared__ float red[1024];
  int k = blockIdx.x, tid = threadIdx.x;
  const float* row = logits + (size_t)k * VPAD;
  float m = -1e30f;
  for (int v = tid; v < VOCAB; v += 1024) m = fmaxf(m, row[v]);
  red[tid] = m; __syncthreads();
  for (int s = 512; s > 0; s >>= 1) { if (tid < s) red[tid] = fmaxf(red[tid], red[tid + s]); __syncthreads(); }
  float mx = red[0]; __syncthreads();
  float sum = 0.f;
  for (int v = tid; v < VOCAB; v += 1024) sum += __expf(row[v] - mx);
  red[tid] = sum; __syncthreads();
  for (int s = 512; s > 0; s >>= 1) { if (tid < s) red[tid] += red[tid + s]; __syncthreads(); }
  if (tid == 0) { rowmax[k] = mx; rowsum[k] = red[0]; }
}

__global__ void k_topk(const float* logits, const float* rowmax, const float* rowsum,
                       const float* probs, const int* fin, float* top_p, int* top_idx) {
  __shared__ float sv[4096]; __shared__ int si[4096];
  __shared__ float rv[1024]; __shared__ int rc[1024]; __shared__ int rp[1024];
  int tid = threadIdx.x;
  float bv4[4]; int bi4[4];
#pragma unroll
  for (int j = 0; j < 4; ++j) { bv4[j] = -1e30f; bi4[j] = 0x7fffffff; }
  for (int k = 0; k < KBEAM; ++k) {
    float pk = probs[k]; int fk = fin[k];
    float mx = rowmax[k]; float inv = pk / rowsum[k];
    const float* row = logits + (size_t)k * VPAD;
    for (int v = tid; v < VOCAB; v += 1024) {
      float val = fk ? ((v == EOS_T) ? pk : 0.0f) : (inv * __expf(row[v] - mx));
      int ci = k * VOCAB + v;
      if (val > bv4[3] || (val == bv4[3] && ci < bi4[3])) {
        bv4[3] = val; bi4[3] = ci;
#pragma unroll
        for (int j = 3; j > 0; --j)
          if (bv4[j] > bv4[j - 1] || (bv4[j] == bv4[j - 1] && bi4[j] < bi4[j - 1])) {
            float tv = bv4[j]; bv4[j] = bv4[j - 1]; bv4[j - 1] = tv;
            int ti = bi4[j]; bi4[j] = bi4[j - 1]; bi4[j - 1] = ti;
          }
      }
    }
  }
#pragma unroll
  for (int j = 0; j < 4; ++j) { sv[tid * 4 + j] = bv4[j]; si[tid * 4 + j] = bi4[j]; }
  __syncthreads();
  for (int r = 0; r < KBEAM; ++r) {
    float bv = -1e30f; int bc = 0x7fffffff, bp = tid * 4;
#pragma unroll
    for (int j = 0; j < 4; ++j) {
      int p = tid * 4 + j;
      if (sv[p] > bv || (sv[p] == bv && si[p] < bc)) { bv = sv[p]; bc = si[p]; bp = p; }
    }
    rv[tid] = bv; rc[tid] = bc; rp[tid] = bp;
    __syncthreads();
    for (int s = 512; s > 0; s >>= 1) {
      if (tid < s && (rv[tid + s] > rv[tid] || (rv[tid + s] == rv[tid] && rc[tid + s] < rc[tid]))) {
        rv[tid] = rv[tid + s]; rc[tid] = rc[tid + s]; rp[tid] = rp[tid + s];
      }
      __syncthreads();
    }
    if (tid == 0) { top_p[r] = rv[0]; top_idx[r] = rc[0]; sv[rp[0]] = -2e30f; }
    __syncthreads();
  }
}

__global__ void k_beam_update(const float* top_p, const int* top_idx,
                              const int* tok_in, const int* fin_in,
                              const float* h_cell, const float* c_cell,
                              int* tok_out, int* fin_out, int* last_out, float* probs_out,
                              float* h_out, float* c_out, int t) {
  int tid = threadIdx.x;
  if (tid < KBEAM) {
    int idx = top_idx[tid];
    int parent = idx / VOCAB, tok = idx % VOCAB;
    last_out[tid] = tok;
    probs_out[tid] = top_p[tid];
    fin_out[tid] = (fin_in[parent] || tok == EOS_T) ? 1 : 0;
  }
  for (int i = tid; i < KBEAM * (MAXLEN + 1); i += blockDim.x) {
    int j = i / (MAXLEN + 1), pos = i % (MAXLEN + 1);
    int idx = top_idx[j]; int parent = idx / VOCAB; int tok = idx % VOCAB;
    tok_out[i] = (pos == t + 1) ? tok : tok_in[parent * (MAXLEN + 1) + pos];
  }
  for (int i = tid; i < 2 * KBEAM * DDIM; i += blockDim.x) {
    if (i < KBEAM * DDIM) {
      int j = i >> 10, u = i & 1023; int parent = top_idx[j] / VOCAB;
      h_out[i] = h_cell[parent * DDIM + u];
    } else {
      int i2 = i - KBEAM * DDIM, j = i2 >> 10, u = i2 & 1023;
      int parent = top_idx[j] / VOCAB;
      c_out[i2] = c_cell[parent * DDIM + u];
    }
  }
}

__global__ void k_finalize(const int* tokens, const float* probs, float* out) {
  int tid = threadIdx.x;
  int best = 0; float bp = probs[0];
  for (int k = 1; k < KBEAM; ++k) { float p = probs[k]; if (p > bp) { bp = p; best = k; } }
  if (tid < MAXLEN + 1) out[tid] = (float)tokens[best * (MAXLEN + 1) + tid];
  else if (tid < MAXLEN + 1 + KBEAM) out[tid] = probs[tid - (MAXLEN + 1)];
}

// ------------------------------- host driver -------------------------------
extern "C" void kernel_launch(void* const* d_in, const int* in_sizes, int n_in,
                              void* d_out, int out_size, void* d_ws, size_t ws_size,
                              hipStream_t stream) {
  (void)in_sizes; (void)n_in; (void)out_size; (void)ws_size;
  const int*   sentence  = (const int*)  d_in[0];
  const float* enc_emb   = (const float*)d_in[1];
  const float* enc_Wih_f = (const float*)d_in[2];
  const float* enc_Whh_f = (const float*)d_in[3];
  const float* enc_b_f   = (const float*)d_in[4];
  const float* enc_Wih_b = (const float*)d_in[5];
  const float* enc_Whh_b = (const float*)d_in[6];
  const float* enc_b_b   = (const float*)d_in[7];
  const float* dec_emb   = (const float*)d_in[8];
  const float* dec_Wih   = (const float*)d_in[9];
  const float* dec_Whh   = (const float*)d_in[10];
  const float* dec_b     = (const float*)d_in[11];
  const float* cat_W     = (const float*)d_in[12];
  const float* cat_b     = (const float*)d_in[13];
  const float* out_W     = (const float*)d_in[14];
  const float* out_b     = (const float*)d_in[15];

  char* ws = (char*)d_ws;
  size_t off = 0;
  auto alloc = [&](size_t bytes) -> void* {
    off = (off + 255) & ~(size_t)255;
    void* p = ws + off; off += bytes; return p;
  };
  unsigned short* xw   = (unsigned short*)alloc((size_t)SLEN * EDIM * 2);
  unsigned short* WihF = (unsigned short*)alloc((size_t)2048 * 1024 * 2);
  unsigned short* WihB = (unsigned short*)alloc((size_t)2048 * 1024 * 2);
  unsigned short* decW = (unsigned short*)alloc((size_t)4096 * 2048 * 2);
  unsigned short* catW = (unsigned short*)alloc((size_t)1024 * 2048 * 2);
  unsigned short* outW = (unsigned short*)alloc((size_t)VPAD * 1024 * 2);
  float* Zf  = (float*)alloc((size_t)SLEN * 2048 * 4);
  float* Zb  = (float*)alloc((size_t)SLEN * 2048 * 4);
  float* hfF = (float*)alloc(HDIM * 4);
  float* hfB = (float*)alloc(HDIM * 4);
  unsigned short* Ag   = (unsigned short*)alloc((size_t)16 * 2048 * 2);
  unsigned short* Af   = (unsigned short*)alloc((size_t)16 * 2048 * 2);
  unsigned short* Aout = (unsigned short*)alloc((size_t)16 * 1024 * 2);
  float* zbuf   = (float*)alloc((size_t)KBEAM * 4096 * 4);
  float* embf   = (float*)alloc((size_t)KBEAM * 1024 * 4);
  float* h_cell = (float*)alloc((size_t)KBEAM * DDIM * 4);
  float* c_cell = (float*)alloc((size_t)KBEAM * DDIM * 4);
  float* logits = (float*)alloc((size_t)KBEAM * VPAD * 4);
  float* rowmax = (float*)alloc(KBEAM * 4);
  float* rowsum = (float*)alloc(KBEAM * 4);
  float* top_p  = (float*)alloc(KBEAM * 4);
  int*   top_ix = (int*)  alloc(KBEAM * 4);
  float* h_st[2]; float* c_st[2]; int* tok[2]; int* last[2]; int* fin[2]; float* probs[2];
  for (int s = 0; s < 2; ++s) {
    h_st[s]  = (float*)alloc((size_t)KBEAM * DDIM * 4);
    c_st[s]  = (float*)alloc((size_t)KBEAM * DDIM * 4);
    tok[s]   = (int*)  alloc((size_t)KBEAM * (MAXLEN + 1) * 4);
    last[s]  = (int*)  alloc(KBEAM * 4);
    fin[s]   = (int*)  alloc(KBEAM * 4);
    probs[s] = (float*)alloc(KBEAM * 4);
  }

  // prep: bf16 conversions + embedding gather
  k_conv      <<<2048, 256, 0, stream>>>(enc_Wih_f, WihF, (size_t)2048 * 1024);
  k_conv      <<<2048, 256, 0, stream>>>(enc_Wih_b, WihB, (size_t)2048 * 1024);
  k_conv      <<<2048, 256, 0, stream>>>(cat_W,     catW, (size_t)1024 * 2048);
  k_build_decW<<<8192, 256, 0, stream>>>(dec_Wih, dec_Whh, decW);
  k_build_outW<<<16384, 256, 0, stream>>>(out_W, outW);
  k_gather_x  <<<256, 256, 0, stream>>>(sentence, enc_emb, xw);

  // encoder
  k_enc_gemm<<<dim3(64, 2), 256, 0, stream>>>(xw, WihF, WihB, enc_b_f, enc_b_b, Zf, Zb);
  k_enc_scan<<<2, 512, 0, stream>>>(Zf, Zb, enc_Whh_f, enc_Whh_b, hfF, hfB);
  k_init_dec<<<16, 256, 0, stream>>>(hfF, hfB, h_st[0], c_st[0], tok[0], last[0], fin[0], probs[0]);

  // decoder beam-search steps
  for (int t = 0; t < MAXLEN; ++t) {
    int s = t & 1, d = (t + 1) & 1;
    k_stage_gates<<<32, 256, 0, stream>>>(last[s], dec_emb, h_st[s], Ag, embf);
    k_gemm_gates <<<32, 256, 0, stream>>>(Ag, decW, dec_b, zbuf);
    k_dec_cell   <<<16, 256, 0, stream>>>(zbuf, c_st[s], h_cell, c_cell);
    k_stage_feat <<<32, 256, 0, stream>>>(h_cell, embf, Af);
    k_gemm_cat   <<<8, 256, 0, stream>>>(Af, catW, cat_b, Aout);
    k_gemm_logits<<<393, 256, 0, stream>>>(Aout, outW, out_b, logits);
    k_softmax_red<<<KBEAM, 1024, 0, stream>>>(logits, rowmax, rowsum);
    k_topk       <<<1, 1024, 0, stream>>>(logits, rowmax, rowsum, probs[s], fin[s], top_p, top_ix);
    k_beam_update<<<1, 256, 0, stream>>>(top_p, top_ix, tok[s], fin[s], h_cell, c_cell,
                                         tok[d], fin[d], last[d], probs[d], h_st[d], c_st[d], t);
  }
  k_finalize<<<1, 32, 0, stream>>>(tok[0], probs[0], (float*)d_out);
}
